// FlashAttention_31001073943407
// MI455X (gfx1250) — matrix operand order repaired and checked by reading, not compiled, not run on hardware
//
#include <hip/hip_runtime.h>

typedef __bf16 bf16;
typedef __attribute__((ext_vector_type(16))) __bf16 v16bf;
typedef __attribute__((ext_vector_type(8)))  __bf16 v8bf;
typedef __attribute__((ext_vector_type(4)))  __bf16 v4bf;
typedef __attribute__((ext_vector_type(8)))  float  v8f;

#define S_LEN 4096
#define NHEAD 16
#define HDIM  128
#define BM    128      // query rows per block (8 waves x 16)
#define BN    128      // key rows per tile
#define KSTR  136      // Kb row stride in bf16 elems (272B, 16B aligned)
#define VSTR  136      // Vb row stride
#define PSTR  136      // Pb row stride

__device__ __forceinline__ v16bf cat8(v8bf lo, v8bf hi) {
  v16bf r;
#pragma unroll
  for (int i = 0; i < 8; ++i) { r[i] = lo[i]; r[i + 8] = hi[i]; }
  return r;
}

// K tile: fp32 global -> bf16 LDS, row-major [k][d]
__device__ __forceinline__ void load_K(const float* __restrict__ K, int head,
                                       int kbase, int tid, bf16* kb) {
#pragma unroll
  for (int i = 0; i < 16; ++i) {
    const int e   = tid + i * 256;      // 0..4095 float4 slots
    const int row = e >> 5;             // key row 0..127
    const int c4  = (e & 31) << 2;      // d offset 0..124
    const size_t g = ((size_t)(kbase + row) * NHEAD + head) * HDIM + c4;
    const float4 kv = *(const float4*)(K + g);
    v4bf t;
    t[0] = (bf16)kv.x; t[1] = (bf16)kv.y; t[2] = (bf16)kv.z; t[3] = (bf16)kv.w;
    *(v4bf*)(kb + row * KSTR + c4) = t;
  }
}

// V tile: fp32 global -> bf16 LDS, transposed [d][k]
__device__ __forceinline__ void load_V(const float* __restrict__ V, int head,
                                       int kbase, int tid, bf16* vb) {
#pragma unroll
  for (int i = 0; i < 16; ++i) {
    const int e   = tid + i * 256;
    const int row = e >> 5;
    const int c4  = (e & 31) << 2;
    const size_t g = ((size_t)(kbase + row) * NHEAD + head) * HDIM + c4;
    const float4 vv = *(const float4*)(V + g);
    vb[(c4 + 0) * VSTR + row] = (bf16)vv.x;
    vb[(c4 + 1) * VSTR + row] = (bf16)vv.y;
    vb[(c4 + 2) * VSTR + row] = (bf16)vv.z;
    vb[(c4 + 3) * VSTR + row] = (bf16)vv.w;
  }
}

__global__ __launch_bounds__(256, 1)
void FlashAttention_31001073943407_kernel(const float* __restrict__ Q,
                                          const float* __restrict__ K,
                                          const float* __restrict__ V,
                                          float* __restrict__ O) {
  __shared__ __align__(16) bf16 Kb[2][BN * KSTR];     // 68 KB, double-buffered
  __shared__ __align__(16) bf16 Vb[HDIM * VSTR];      // 34 KB, single-buffered
  __shared__ __align__(16) bf16 Pb[8 * 16 * PSTR];    // 34 KB per-wave P^T scratch

  const int tid  = threadIdx.x;
  const int lane = tid & 31;
  const int wave = tid >> 5;
  const int lh   = lane >> 4;    // which 16-lane half
  const int ln   = lane & 15;

  const int head = blockIdx.x & (NHEAD - 1);
  const int qt   = blockIdx.x >> 4;           // 0..31
  const int q0w  = qt * BM + wave * 16;       // first query row of this wave
  const int qcol = q0w + ln;                  // this lane's query column (S^T)

  const float scale = 0.08838834764831845f;   // 1/sqrt(128)

  // ---- Q as B-fragments: lane ln = column q, 16 contiguous d per chunk ----
  v16bf qf[4];
  {
    const float* qp = Q + ((size_t)qcol * NHEAD + head) * HDIM;
#pragma unroll
    for (int c = 0; c < 4; ++c) {
      const int d0 = c * 32 + lh * 16;        // K {0..15} (lanes<16) / {16..31}
      float4 x0 = *(const float4*)(qp + d0);
      float4 x1 = *(const float4*)(qp + d0 + 4);
      float4 x2 = *(const float4*)(qp + d0 + 8);
      float4 x3 = *(const float4*)(qp + d0 + 12);
      v16bf f;
      f[0]  = (bf16)(x0.x * scale); f[1]  = (bf16)(x0.y * scale);
      f[2]  = (bf16)(x0.z * scale); f[3]  = (bf16)(x0.w * scale);
      f[4]  = (bf16)(x1.x * scale); f[5]  = (bf16)(x1.y * scale);
      f[6]  = (bf16)(x1.z * scale); f[7]  = (bf16)(x1.w * scale);
      f[8]  = (bf16)(x2.x * scale); f[9]  = (bf16)(x2.y * scale);
      f[10] = (bf16)(x2.z * scale); f[11] = (bf16)(x2.w * scale);
      f[12] = (bf16)(x3.x * scale); f[13] = (bf16)(x3.y * scale);
      f[14] = (bf16)(x3.z * scale); f[15] = (bf16)(x3.w * scale);
      qf[c] = f;
    }
  }

  // per-lane online-softmax state for query column qcol
  float mrow = -3.0e38f;
  float lrow = 0.0f;
  v8f acc[8];                                 // O (std layout): row q, col d
#pragma unroll
  for (int nd = 0; nd < 8; ++nd)
#pragma unroll
    for (int r = 0; r < 8; ++r) acc[nd][r] = 0.0f;

  const int ktmax = qt;                 // causal: K tiles 0..qt
  const int wbase = wave * 16 * PSTR;

  load_K(K, head, 0, tid, Kb[0]);       // prefetch first K tile
  __syncthreads();

  for (int kt = 0; kt <= ktmax; ++kt) {
    const int cur   = kt & 1;
    const int kbase = kt * BN;

    // V for this tile (consumed late, after the mid barrier)
    load_V(V, head, kbase, tid, Vb);
    // prefetch next K tile into the other buffer
    if (kt < ktmax) load_K(K, head, kbase + BN, tid, Kb[cur ^ 1]);

    const bf16* kbuf = Kb[cur];

    // ---- S^T = K @ Q^T : 8 k-subtiles x 4 d-chunks of WMMA ----
    // element (M = k_local = nt*16 + r + lh*8, N = q = ln)
    v8f sf[8];
#pragma unroll
    for (int nt = 0; nt < 8; ++nt) {
      v8f c;
#pragma unroll
      for (int r = 0; r < 8; ++r) c[r] = 0.0f;
#pragma unroll
      for (int ch = 0; ch < 4; ++ch) {
        const bf16* ap = kbuf + (nt * 16 + ln) * KSTR + ch * 32 + lh * 16;
        v16bf a = cat8(*(const v8bf*)ap, *(const v8bf*)(ap + 8));
        c = __builtin_amdgcn_wmma_f32_16x16x32_bf16(false, a, false, qf[ch],
                                                    (short)0, c, false, false);
      }
      sf[nt] = c;
    }

    // ---- causal mask + per-lane max, one cross-half merge ----
    const bool edge = (kt == ktmax);    // only the diagonal tile needs masking
    float mx = -3.0e38f;
#pragma unroll
    for (int nt = 0; nt < 8; ++nt)
#pragma unroll
      for (int r = 0; r < 8; ++r) {
        float s = sf[nt][r];
        if (edge && (kbase + nt * 16 + r + lh * 8) > qcol) {
          s = -1.0e10f; sf[nt][r] = s;
        }
        mx = fmaxf(mx, s);
      }
    mx = fmaxf(mx, __shfl_xor(mx, 16, 32));
    const float mnew = fmaxf(mrow, mx);

    // ---- rescale acc/l only when the running max grew (uniform branch) ----
    if (__any(mnew > mrow)) {
      const float al = __expf(mrow - mnew);   // per-lane alpha (q = qcol)
      lrow *= al;
      float ar[8];                            // broadcast to per-row alpha
#pragma unroll
      for (int r = 0; r < 8; ++r) ar[r] = __shfl(al, r + lh * 8, 32);
#pragma unroll
      for (int nd = 0; nd < 8; ++nd)
#pragma unroll
        for (int r = 0; r < 8; ++r) acc[nd][r] *= ar[r];
    }
    mrow = mnew;

    // ---- P = exp(S - m): per-lane sum, packed bf16 b128 stores of P^T ----
    float sum = 0.0f;
#pragma unroll
    for (int nt = 0; nt < 8; ++nt) {
      v8bf p8;
#pragma unroll
      for (int r = 0; r < 8; ++r) {
        const float p = __expf(sf[nt][r] - mnew);
        sum += p;
        p8[r] = (bf16)p;
      }
      // P[q=ln][k = nt*16 + lh*8 + 0..7], 8 contiguous bf16
      *(v8bf*)(Pb + wbase + ln * PSTR + nt * 16 + lh * 8) = p8;
    }
    sum += __shfl_xor(sum, 16, 32);
    lrow += sum;

    __syncthreads();   // publishes P (own wave) and V tile (all waves)

    // ---- O += P @ V : 4 k-chunks x 8 d-tiles of WMMA ----
#pragma unroll
    for (int ch = 0; ch < 4; ++ch) {
      const bf16* pp = Pb + wbase + ln * PSTR + ch * 32 + lh * 8;
      const v16bf af = cat8(*(const v8bf*)pp, *(const v8bf*)(pp + 16));
#pragma unroll
      for (int nd = 0; nd < 8; ++nd) {
        const bf16* vp = Vb + (nd * 16 + ln) * VSTR + ch * 32 + lh * 8;
        v16bf b = cat8(*(const v8bf*)vp, *(const v8bf*)(vp + 16));
        acc[nd] = __builtin_amdgcn_wmma_f32_16x16x32_bf16(false, af, false, b,
                                                          (short)0, acc[nd],
                                                          false, false);
      }
    }

    __syncthreads();   // all reads of Kb[cur]/Vb done before next tile's writes
  }

  // ---- epilogue: O = acc / (l + 1e-8), fp32 store [B,S,H,D] ----
  float inv[8];
#pragma unroll
  for (int r = 0; r < 8; ++r) {
    const float lr = __shfl(lrow, r + lh * 8, 32);   // l for row q0w + r + lh*8
    inv[r] = 1.0f / (lr + 1e-8f);
  }
#pragma unroll
  for (int nd = 0; nd < 8; ++nd) {
    const int d = nd * 16 + ln;
#pragma unroll
    for (int r = 0; r < 8; ++r) {
      const int qrow = q0w + r + lh * 8;
      O[((size_t)qrow * NHEAD + head) * HDIM + d] = acc[nd][r] * inv[r];
    }
  }
}

extern "C" void kernel_launch(void* const* d_in, const int* in_sizes, int n_in,
                              void* d_out, int out_size, void* d_ws, size_t ws_size,
                              hipStream_t stream) {
  (void)in_sizes; (void)n_in; (void)out_size; (void)d_ws; (void)ws_size;
  const float* Q = (const float*)d_in[0];
  const float* K = (const float*)d_in[1];
  const float* V = (const float*)d_in[2];
  float* O = (float*)d_out;
  dim3 grid(NHEAD * (S_LEN / BM));   // 512 blocks
  dim3 block(256);                   // 8 wave32
  FlashAttention_31001073943407_kernel<<<grid, block, 0, stream>>>(Q, K, V, O);
}
